// MultiHeadedAttentionGoogle_3075196584184
// MI455X (gfx1250) — compile-verified, matrix operand order checked by reading, not run on hardware
//
#include <hip/hip_runtime.h>
#include <hip/hip_bf16.h>

typedef __attribute__((ext_vector_type(8)))  float  v8f;
typedef __attribute__((ext_vector_type(16))) __bf16 v16bf;
typedef __attribute__((ext_vector_type(8)))  __bf16 v8bf;

#define BATCH  4
#define SEQ    2048
#define DMODEL 1024
#define NHEAD  16
#define DHEAD  64
// M = BATCH*SEQ = 8192 rows for every GEMM, K = N = 1024.

__device__ __forceinline__ __bf16 f2bf(float f) {
    unsigned u = __builtin_bit_cast(unsigned, f);
    u += 0x7FFFu + ((u >> 16) & 1u);          // round-to-nearest-even
    return __builtin_bit_cast(__bf16, (unsigned short)(u >> 16));
}

// Load one bf16 WMMA fragment slice for this lane.
// Caller passes p = row_base + k0 + half*8 ; we grab bf16[0..7] and bf16[16..23].
__device__ __forceinline__ v16bf ldfrag(const __bf16* p) {
    v8bf lo = *reinterpret_cast<const v8bf*>(p);
    v8bf hi = *reinterpret_cast<const v8bf*>(p + 16);
    return __builtin_shufflevector(lo, hi,
        0,1,2,3,4,5,6,7,8,9,10,11,12,13,14,15);
}

// ---------------- elementwise converts ----------------
__global__ __launch_bounds__(256) void cvt_f32_bf16(const float* __restrict__ in,
                                                    __bf16* __restrict__ out, int n) {
    int i = blockIdx.x * 256 + threadIdx.x;
    if (i < n) out[i] = f2bf(in[i]);
}

__global__ __launch_bounds__(256) void mask_to_bias(const int* __restrict__ mask,
                                                    float* __restrict__ mb, int n) {
    int i = blockIdx.x * 256 + threadIdx.x;
    if (i < n) mb[i] = (mask[i] == 0) ? -1.0e9f : 0.0f;
}

// ---------------- bf16 GEMM: Y = X @ W^T + b ----------------
// X: [8192,1024] bf16 row-major, W: [1024,1024] bf16 row-major (row n = output col n).
// MODE 0: bf16 out -> [B,H,S,DK]   (Q,K)
// MODE 1: bf16 out -> [B,H,DK,S]   (V transposed for the PV stage)
// MODE 2: f32  out -> [M,N]        (final projection -> d_out)
template<int MODE>
__global__ __launch_bounds__(256) void gemm_bf16(const __bf16* __restrict__ A,
                                                 const __bf16* __restrict__ W,
                                                 const float*  __restrict__ bias,
                                                 __bf16* __restrict__ outB,
                                                 float*  __restrict__ outF) {
    const int lane = threadIdx.x & 31, wave = threadIdx.x >> 5;
    const int l15 = lane & 15, half = lane >> 4;
    const int wm = blockIdx.y * 128 + (wave & 3) * 32;   // 4 waves along M
    const int wn = blockIdx.x * 128 + (wave >> 2) * 64;  // 2 waves along N

    const __bf16* Ap = A + (size_t)(wm + l15) * DMODEL + half * 8;
    const __bf16* Wp = W + (size_t)(wn + l15) * DMODEL + half * 8;

    v8f acc[2][4];
#pragma unroll
    for (int i = 0; i < 2; ++i)
#pragma unroll
        for (int j = 0; j < 4; ++j) acc[i][j] = (v8f){0,0,0,0,0,0,0,0};

    for (int k0 = 0; k0 < DMODEL; k0 += 32) {
        __builtin_prefetch(Ap + k0 + 64, 0, 1);          // global_prefetch_b8
        __builtin_prefetch(Wp + k0 + 64, 0, 1);
        v16bf af[2], bf[4];
#pragma unroll
        for (int i = 0; i < 2; ++i)
            af[i] = ldfrag(Ap + (size_t)(16 * i) * DMODEL + k0);
#pragma unroll
        for (int j = 0; j < 4; ++j)
            bf[j] = ldfrag(Wp + (size_t)(16 * j) * DMODEL + k0);
#pragma unroll
        for (int i = 0; i < 2; ++i)
#pragma unroll
            for (int j = 0; j < 4; ++j)
                acc[i][j] = __builtin_amdgcn_wmma_f32_16x16x32_bf16(
                    false, af[i], false, bf[j], (short)0, acc[i][j], false, false);
    }

    // ---- epilogue: per-wave constant bases, straight-line stores ----
    float bvj[4];
#pragma unroll
    for (int j = 0; j < 4; ++j) bvj[j] = bias[wn + 16 * j + l15];

    const int bb = wm >> 11;          // batch  (constant per wave: 128 | 2048)
    const int hh = wn >> 6;           // head   (constant per wave: 64-aligned wn)
    const int ms = wm & 2047;         // row within batch
    const int rowoff = 8 * half;      // C layout: M = r + 8*half

    if (MODE == 0) {
        __bf16* base = outB + ((size_t)(bb * NHEAD + hh) * SEQ + ms) * DHEAD + l15;
#pragma unroll
        for (int i = 0; i < 2; ++i)
#pragma unroll
            for (int j = 0; j < 4; ++j)
#pragma unroll
                for (int r = 0; r < 8; ++r)
                    base[(size_t)(16 * i + rowoff + r) * DHEAD + 16 * j] =
                        f2bf(acc[i][j][r] + bvj[j]);
    } else if (MODE == 1) {
        __bf16* base = outB + ((size_t)(bb * NHEAD + hh) * DHEAD + l15) * SEQ + ms;
#pragma unroll
        for (int i = 0; i < 2; ++i)
#pragma unroll
            for (int j = 0; j < 4; ++j)
#pragma unroll
                for (int r = 0; r < 8; ++r)
                    base[(size_t)(16 * j) * SEQ + 16 * i + rowoff + r] =
                        f2bf(acc[i][j][r] + bvj[j]);
    } else {
        float* base = outF + (size_t)wm * DMODEL + wn + l15;
#pragma unroll
        for (int i = 0; i < 2; ++i)
#pragma unroll
            for (int j = 0; j < 4; ++j)
#pragma unroll
                for (int r = 0; r < 8; ++r)
                    base[(size_t)(16 * i + rowoff + r) * DMODEL + 16 * j] =
                        acc[i][j][r] + bvj[j];
    }
}

// ---------------- flash attention (S^T / O^T formulation) ----------------
// grid = B*H*(S/128), 8 waves, each wave owns 16 queries.
__global__ __launch_bounds__(256) void attn_flash(const __bf16* __restrict__ Qh,
                                                  const __bf16* __restrict__ Kh,
                                                  const __bf16* __restrict__ Vth,
                                                  const float*  __restrict__ mb,
                                                  __bf16* __restrict__ ctx) {
    const int lane = threadIdx.x & 31, wave = threadIdx.x >> 5;
    const int l15 = lane & 15, half = lane >> 4;
    const int qt  = blockIdx.x & 15;         // 16 query tiles of 128
    const int bhi = blockIdx.x >> 4;         // b*H + h
    const int b   = bhi >> 4, h = bhi & 15;
    const int q0  = qt * 128 + wave * 16;

    const __bf16* Qp  = Qh  + (size_t)bhi * SEQ * DHEAD;   // [S][64]
    const __bf16* Kp  = Kh  + ((size_t)bhi * SEQ + l15) * DHEAD + half * 8;
    const __bf16* Vp  = Vth + ((size_t)bhi * DHEAD + l15) * SEQ + half * 8;
    const float*  mbb = mb  + (size_t)b * SEQ + 16 * half; // keys this lane sees
    // (mbb[kb + 16u - 8*half]... handled below with explicit index)
    const float*  mrow = mb + (size_t)b * SEQ + 8 * half;

    // Q^T as B-fragments (lane = query column), dk chunks of 32.
    v16bf bq[2];
#pragma unroll
    for (int kk = 0; kk < 2; ++kk)
        bq[kk] = ldfrag(Qp + (size_t)(q0 + l15) * DHEAD + 32 * kk + half * 8);

    v8f vacc[4];                                   // O^T: 64 dk rows x 16 queries
#pragma unroll
    for (int t = 0; t < 4; ++t) vacc[t] = (v8f){0,0,0,0,0,0,0,0};
    float m_run = -1.0e30f, l_run = 0.0f;

    const v8f zc = (v8f){0,0,0,0,0,0,0,0};

    for (int kb = 0; kb < SEQ; kb += 32) {
        __builtin_prefetch(Kp + (size_t)(kb + 32) * DHEAD, 0, 1);
        __builtin_prefetch(Vp + kb + 32, 0, 1);
        // S^T = K_tile @ Q^T : rows = keys (regs), cols = queries (lanes)
        v8f st[2];
#pragma unroll
        for (int u = 0; u < 2; ++u) {
            v16bf ka0 = ldfrag(Kp + (size_t)(kb + 16 * u) * DHEAD);
            v16bf ka1 = ldfrag(Kp + (size_t)(kb + 16 * u) * DHEAD + 32);
            v8f s = __builtin_amdgcn_wmma_f32_16x16x32_bf16(
                        false, ka0, false, bq[0], (short)0, zc, false, false);
            st[u] = __builtin_amdgcn_wmma_f32_16x16x32_bf16(
                        false, ka1, false, bq[1], (short)0, s, false, false);
        }
        // scale + mask bias, tile max (keys this lane sees: kb+16u+8*half+r)
        float tm = -1.0e30f;
#pragma unroll
        for (int u = 0; u < 2; ++u)
#pragma unroll
            for (int r = 0; r < 8; ++r) {
                float s = st[u][r] * 0.125f + mrow[kb + 16 * u + r];
                st[u][r] = s;
                tm = fmaxf(tm, s);
            }
        tm = fmaxf(tm, __shfl_xor(tm, 16, 32));     // pair lanes see same query
        const float m_new = fmaxf(m_run, tm);
        const float scale = __expf(m_run - m_new);
        float tsum = 0.0f;
#pragma unroll
        for (int u = 0; u < 2; ++u)
#pragma unroll
            for (int r = 0; r < 8; ++r) {
                float p = __expf(st[u][r] - m_new);
                st[u][r] = p;
                tsum += p;
            }
        tsum += __shfl_xor(tsum, 16, 32);
        l_run = l_run * scale + tsum;
        m_run = m_new;

        // P^T C-layout is exactly the B-fragment layout for the next WMMA.
        v16bf bp;
#pragma unroll
        for (int r = 0; r < 8; ++r) {
            bp[r]     = f2bf(st[0][r]);
            bp[8 + r] = f2bf(st[1][r]);
        }
        // O^T += V^T_tile @ P^T   (A rows = dk, contiguous keys from V^T layout)
#pragma unroll
        for (int t = 0; t < 4; ++t) {
#pragma unroll
            for (int r = 0; r < 8; ++r) vacc[t][r] *= scale;
            v16bf va = ldfrag(Vp + (size_t)(16 * t) * SEQ + kb);
            vacc[t] = __builtin_amdgcn_wmma_f32_16x16x32_bf16(
                          false, va, false, bp, (short)0, vacc[t], false, false);
        }
    }

    const float inv = 1.0f / l_run;
    // ctx[b, q, h*64 + dk], dk = 16t + 8*half + r, q = q0 + l15
    __bf16* cbase = ctx + (size_t)(b * SEQ + q0 + l15) * DMODEL + h * DHEAD + 8 * half;
#pragma unroll
    for (int t = 0; t < 4; ++t)
#pragma unroll
        for (int r = 0; r < 8; ++r)
            cbase[16 * t + r] = f2bf(vacc[t][r] * inv);
}

// ---------------- launch ----------------
extern "C" void kernel_launch(void* const* d_in, const int* in_sizes, int n_in,
                              void* d_out, int out_size, void* d_ws, size_t ws_size,
                              hipStream_t stream) {
    const float* query = (const float*)d_in[0];
    const float* key   = (const float*)d_in[1];
    const float* value = (const float*)d_in[2];
    const int*   mask  = (const int*)d_in[3];
    const float* Wq = (const float*)d_in[4];
    const float* bq = (const float*)d_in[5];
    const float* Wk = (const float*)d_in[6];
    const float* bk = (const float*)d_in[7];
    const float* Wv = (const float*)d_in[8];
    const float* bv = (const float*)d_in[9];
    const float* Wo = (const float*)d_in[10];
    const float* bo = (const float*)d_in[11];
    float* out = (float*)d_out;

    char* wsp = (char*)d_ws;
    size_t off = 0;
    auto alloc = [&](size_t bytes) -> char* {
        char* p = wsp + off;
        off += (bytes + 255) & ~(size_t)255;
        return p;
    };

    const size_t NX = (size_t)BATCH * SEQ * DMODEL;   // 8,388,608
    const size_t NW = (size_t)DMODEL * DMODEL;        // 1,048,576

    __bf16* Xq  = (__bf16*)alloc(NX * 2);
    __bf16* Xk  = (__bf16*)alloc(NX * 2);
    __bf16* Xv  = (__bf16*)alloc(NX * 2);
    __bf16* Wqb = (__bf16*)alloc(NW * 2);
    __bf16* Wkb = (__bf16*)alloc(NW * 2);
    __bf16* Wvb = (__bf16*)alloc(NW * 2);
    __bf16* Wob = (__bf16*)alloc(NW * 2);
    __bf16* Qh  = (__bf16*)alloc(NX * 2);             // [B,H,S,DK]
    __bf16* Kh  = (__bf16*)alloc(NX * 2);             // [B,H,S,DK]
    __bf16* Vth = (__bf16*)alloc(NX * 2);             // [B,H,DK,S]
    __bf16* ctx = (__bf16*)alloc(NX * 2);             // [B,S,D]
    float*  mb  = (float*)alloc((size_t)BATCH * SEQ * 4);

    const int nBlkX = (int)((NX + 255) / 256);
    const int nBlkW = (int)((NW + 255) / 256);
    cvt_f32_bf16<<<nBlkX, 256, 0, stream>>>(query, Xq, (int)NX);
    cvt_f32_bf16<<<nBlkX, 256, 0, stream>>>(key,   Xk, (int)NX);
    cvt_f32_bf16<<<nBlkX, 256, 0, stream>>>(value, Xv, (int)NX);
    cvt_f32_bf16<<<nBlkW, 256, 0, stream>>>(Wq, Wqb, (int)NW);
    cvt_f32_bf16<<<nBlkW, 256, 0, stream>>>(Wk, Wkb, (int)NW);
    cvt_f32_bf16<<<nBlkW, 256, 0, stream>>>(Wv, Wvb, (int)NW);
    cvt_f32_bf16<<<nBlkW, 256, 0, stream>>>(Wo, Wob, (int)NW);
    mask_to_bias<<<(BATCH * SEQ) / 256, 256, 0, stream>>>(mask, mb, BATCH * SEQ);

    dim3 ggrid(DMODEL / 128, (BATCH * SEQ) / 128);    // (8, 64)
    gemm_bf16<0><<<ggrid, 256, 0, stream>>>(Xq, Wqb, bq, Qh,  nullptr);
    gemm_bf16<0><<<ggrid, 256, 0, stream>>>(Xk, Wkb, bk, Kh,  nullptr);
    gemm_bf16<1><<<ggrid, 256, 0, stream>>>(Xv, Wvb, bv, Vth, nullptr);

    attn_flash<<<BATCH * NHEAD * (SEQ / 128), 256, 0, stream>>>(Qh, Kh, Vth, mb, ctx);

    gemm_bf16<2><<<ggrid, 256, 0, stream>>>(ctx, Wob, bo, nullptr, out);
}
